// TransLayer_81595788690061
// MI455X (gfx1250) — matrix operand order repaired and checked by reading, not compiled, not run on hardware
//
#include <hip/hip_runtime.h>
#include <stdint.h>

// ---------------------------------------------------------------------------
// Types for CDNA5 WMMA (wave32): v_wmma_f32_16x16x32_bf16
// ---------------------------------------------------------------------------
typedef __attribute__((ext_vector_type(16))) __bf16 bf16x16;
typedef __attribute__((ext_vector_type(8)))  float  floatx8;

union FragBf {
    bf16x16 f;
    uint4   q[2];   // 32 bytes = 16 bf16
};

__device__ __forceinline__ unsigned short f2bf(float x) {
    unsigned int u = __float_as_uint(x);
    unsigned int r = u + 0x7FFFu + ((u >> 16) & 1u);   // round-to-nearest-even
    return (unsigned short)(r >> 16);
}

// ---------------------------------------------------------------------------
// Generic strided-batched GEMM:  C(f32)[M,N] = A(bf16)[M,K] * B(bf16)
//   bT==0 : B stored row-major [K,N] (ldb = N)
//   bT==1 : B stored row-major [N,K] (ldb = K), used transposed (contract K)
// REQUIREMENTS (all launches below satisfy these):
//   M % 128 == 0, K % 32 == 0, and for bT==1 also N % 128 == 0.
// Block tile 128x128, K-step 32, 256 threads = 8 waves, each wave owns a
// 32x64 sub-tile = 2x4 grid of 16x16 WMMA accumulators.
// ---------------------------------------------------------------------------
#define BM 128
#define BN 128
#define BK 32
#define LDSS 40   // LDS row stride in bf16 elems (multiple of 8, padded)

__global__ __launch_bounds__(256)
void gemm_bf16_kernel(const unsigned short* __restrict__ A, size_t strideA, int lda,
                      const unsigned short* __restrict__ B, size_t strideB, int ldb, int bT,
                      float* __restrict__ C, size_t strideC, int ldc,
                      int M, int N, int K)
{
    __shared__ alignas(16) unsigned short sA[BM * LDSS];
    __shared__ alignas(16) unsigned short sB[BN * LDSS];   // stored [n][k]

    const int batch = blockIdx.z;
    A += (size_t)batch * strideA;
    B += (size_t)batch * strideB;
    C += (size_t)batch * strideC;

    const int m0  = blockIdx.y * BM;
    const int n0  = blockIdx.x * BN;
    const int tid = threadIdx.x;
    const int wave = tid >> 5;
    const int lane = tid & 31;
    const int wr = (wave >> 1) * 32;   // 0,32,64,96
    const int wc = (wave & 1) * 64;    // 0,64

    floatx8 acc[2][4];
    const floatx8 zacc = {0.f,0.f,0.f,0.f,0.f,0.f,0.f,0.f};
    for (int i = 0; i < 2; ++i)
        for (int j = 0; j < 4; ++j) acc[i][j] = zacc;

    for (int k0 = 0; k0 < K; k0 += BK) {
        // ---- prefetch next K-chunk into cache while this one is consumed ----
        if (k0 + BK < K) {
            if (tid < BM)
                __builtin_prefetch(A + (size_t)(m0 + tid) * lda + (k0 + BK), 0, 1);
            if (bT) {
                if (tid < BN)
                    __builtin_prefetch(B + (size_t)(n0 + tid) * ldb + (k0 + BK), 0, 1);
            } else {
                if (tid < BK)
                    __builtin_prefetch(B + (size_t)(k0 + BK + tid) * ldb + n0, 0, 1);
            }
        }

        // ---- stage A tile: BM x BK (vectorized, unguarded: M % 128 == 0) ----
        for (int i = tid; i < BM * (BK / 8); i += 256) {
            int r  = i >> 2;          // BK/8 == 4
            int kc = (i & 3) * 8;
            *(uint4*)(sA + r * LDSS + kc) =
                *(const uint4*)(A + (size_t)(m0 + r) * lda + (k0 + kc));
        }
        // ---- stage B tile as [n][k] ----
        if (bT) {
            // unguarded: bT launches have N % 128 == 0
            for (int i = tid; i < BN * (BK / 8); i += 256) {
                int n  = i >> 2;
                int kc = (i & 3) * 8;
                *(uint4*)(sB + n * LDSS + kc) =
                    *(const uint4*)(B + (size_t)(n0 + n) * ldb + (k0 + kc));
            }
        } else {
            // branchless bounds handling (clamp + select) for N=64 launches
            for (int i = tid; i < BN * BK; i += 256) {
                int n = i & (BN - 1);
                int k = i >> 7;
                int gn = n0 + n;
                int nc = (gn < N) ? gn : (N - 1);
                unsigned short v = B[(size_t)(k0 + k) * ldb + nc];
                sB[n * LDSS + k] = (gn < N) ? v : (unsigned short)0;
            }
        }
        __syncthreads();

        // ---- gather fragments per documented wave32 layouts ----
        const int lrow = lane & 15;
        const int kbA  = (lane < 16) ? 0 : 8;    // A: K {0-7,16-23} / {8-15,24-31}
        const int kbB  = (lane < 16) ? 0 : 8;    // B: K {0-7,16-23} / {8-15,24-31}
        FragBf fa[2], fb[4];
        for (int rt = 0; rt < 2; ++rt) {
            const unsigned short* p = sA + (wr + rt * 16 + lrow) * LDSS;
            fa[rt].q[0] = *(const uint4*)(p + kbA);
            fa[rt].q[1] = *(const uint4*)(p + kbA + 16);
        }
        for (int ct = 0; ct < 4; ++ct) {
            const unsigned short* p = sB + (wc + ct * 16 + lrow) * LDSS;
            fb[ct].q[0] = *(const uint4*)(p + kbB);
            fb[ct].q[1] = *(const uint4*)(p + kbB + 16);
        }
        for (int rt = 0; rt < 2; ++rt)
            for (int ct = 0; ct < 4; ++ct)
                acc[rt][ct] = __builtin_amdgcn_wmma_f32_16x16x32_bf16(
                    false, fa[rt].f, false, fb[ct].f,
                    (short)0, acc[rt][ct], false, false);
        __syncthreads();
    }

    // ---- store: C/D layout (lanes 0-15: M=r, lanes 16-31: M=r+8; N=lane&15) ----
    const int lcol  = lane & 15;
    const int rhalf = (lane < 16) ? 0 : 8;
    for (int rt = 0; rt < 2; ++rt) {
        for (int ct = 0; ct < 4; ++ct) {
            int c = n0 + wc + ct * 16 + lcol;
            if (c >= N) continue;
            int rb = m0 + wr + rt * 16 + rhalf;
#pragma unroll
            for (int r = 0; r < 8; ++r)
                C[(size_t)(rb + r) * ldc + c] = acc[rt][ct][r];
        }
    }
}

// ---------------------------------------------------------------------------
// LayerNorm over 512 columns -> bf16, one block per row
// ---------------------------------------------------------------------------
__global__ __launch_bounds__(256)
void layernorm_kernel(const float* __restrict__ x, const float* __restrict__ gamma,
                      const float* __restrict__ beta, unsigned short* __restrict__ out)
{
    __shared__ float red[256];
    const size_t row = blockIdx.x;
    const float* xr  = x + row * 512;
    const int t = threadIdx.x;
    float v0 = xr[t], v1 = xr[t + 256];
    red[t] = v0 + v1;
    __syncthreads();
    for (int off = 128; off > 0; off >>= 1) {
        if (t < off) red[t] += red[t + off];
        __syncthreads();
    }
    float mu = red[0] * (1.f / 512.f);
    __syncthreads();
    float d0 = v0 - mu, d1 = v1 - mu;
    red[t] = d0 * d0 + d1 * d1;
    __syncthreads();
    for (int off = 128; off > 0; off >>= 1) {
        if (t < off) red[t] += red[t + off];
        __syncthreads();
    }
    float rstd = rsqrtf(red[0] * (1.f / 512.f) + 1e-5f);
    out[row * 512 + t]       = f2bf(d0 * rstd * gamma[t]       + beta[t]);
    out[row * 512 + t + 256] = f2bf(d1 * rstd * gamma[t + 256] + beta[t + 256]);
}

// ---------------------------------------------------------------------------
// Row softmax (f32 in), writes bf16 and optionally f32
// ---------------------------------------------------------------------------
__global__ __launch_bounds__(256)
void softmax_kernel(const float* __restrict__ S, unsigned short* __restrict__ obf,
                    float* __restrict__ of32, int ncols)
{
    __shared__ float red[256];
    const size_t row = blockIdx.x;
    const float* sr  = S + row * (size_t)ncols;
    const int t = threadIdx.x;
    float mx = -3.4e38f;
    for (int c = t; c < ncols; c += 256) mx = fmaxf(mx, sr[c]);
    red[t] = mx;
    __syncthreads();
    for (int off = 128; off > 0; off >>= 1) {
        if (t < off) red[t] = fmaxf(red[t], red[t + off]);
        __syncthreads();
    }
    mx = red[0];
    __syncthreads();
    float sum = 0.f;
    for (int c = t; c < ncols; c += 256) sum += __expf(sr[c] - mx);
    red[t] = sum;
    __syncthreads();
    for (int off = 128; off > 0; off >>= 1) {
        if (t < off) red[t] += red[t + off];
        __syncthreads();
    }
    float inv = 1.f / red[0];
    for (int c = t; c < ncols; c += 256) {
        float p = __expf(sr[c] - mx) * inv;
        obf[row * (size_t)ncols + c] = f2bf(p);
        if (of32) of32[row * (size_t)ncols + c] = p;
    }
}

// ---------------------------------------------------------------------------
// Split heads from qkv f32 [B,N,1536] -> qs/k/v bf16 [BH,N,64] (+ v f32)
// ---------------------------------------------------------------------------
__global__ __launch_bounds__(256)
void reorg_kernel(const float* __restrict__ qkv, unsigned short* __restrict__ qs,
                  unsigned short* __restrict__ kb, unsigned short* __restrict__ vb,
                  float* __restrict__ vf)
{
    size_t idx = (size_t)blockIdx.x * 256 + threadIdx.x;     // B*N*512
    if (idx >= (size_t)4 * 4096 * 512) return;
    int c = idx & 511;
    size_t bn = idx >> 9;
    int n = (int)(bn & 4095);
    int b = (int)(bn >> 12);
    int h = c >> 6, d = c & 63;
    const float* base = qkv + bn * 1536;
    size_t o = (((size_t)(b * 8 + h) * 4096) + n) * 64 + d;
    qs[o] = f2bf(base[c] * 0.125f);            // SCALE = 64^-0.5
    kb[o] = f2bf(base[512 + c]);
    float v = base[1024 + c];
    vb[o] = f2bf(v);
    vf[o] = v;
}

// ---------------------------------------------------------------------------
// Landmark means over groups of 16 tokens (q scaled), from f32 qkv
// ---------------------------------------------------------------------------
__global__ __launch_bounds__(64)
void landmarks_kernel(const float* __restrict__ qkv,
                      unsigned short* __restrict__ qlb, unsigned short* __restrict__ klb)
{
    int j  = blockIdx.x;    // landmark 0..255
    int bh = blockIdx.y;    // 0..31
    int d  = threadIdx.x;   // 0..63
    int b = bh >> 3, h = bh & 7;
    float sq = 0.f, sk = 0.f;
    size_t base = (((size_t)b * 4096) + (size_t)j * 16) * 1536 + h * 64 + d;
    for (int t = 0; t < 16; ++t) {
        sq += qkv[base + (size_t)t * 1536];
        sk += qkv[base + (size_t)t * 1536 + 512];
    }
    size_t o = ((size_t)bh * 256 + j) * 64 + d;
    qlb[o] = f2bf(sq * (0.125f / 16.f));
    klb[o] = f2bf(sk * (1.f / 16.f));
}

// ---------------------------------------------------------------------------
// pinv helpers (all on 32 batches of 256x256)
// ---------------------------------------------------------------------------
__global__ void init_scalars_kernel(float* s) { if (threadIdx.x < 2) s[threadIdx.x] = 0.f; }

__global__ __launch_bounds__(256)
void rowsum_max_kernel(const float* __restrict__ a, float* scal)  // col = max row abs-sum
{
    __shared__ float red[256];
    const float* ar = a + (size_t)blockIdx.x * 256;
    red[threadIdx.x] = fabsf(ar[threadIdx.x]);
    __syncthreads();
    for (int off = 128; off > 0; off >>= 1) {
        if (threadIdx.x < off) red[threadIdx.x] += red[threadIdx.x + off];
        __syncthreads();
    }
    if (threadIdx.x == 0) atomicMax((int*)scal, __float_as_int(red[0]));
}

__global__ __launch_bounds__(256)
void colsum_max_kernel(const float* __restrict__ a, float* scal)  // row = max col abs-sum
{
    __shared__ float red[256];
    int bh = blockIdx.x >> 8, j = blockIdx.x & 255;
    const float* ab = a + (size_t)bh * 65536;
    red[threadIdx.x] = fabsf(ab[(size_t)threadIdx.x * 256 + j]);
    __syncthreads();
    for (int off = 128; off > 0; off >>= 1) {
        if (threadIdx.x < off) red[threadIdx.x] += red[threadIdx.x + off];
        __syncthreads();
    }
    if (threadIdx.x == 0) atomicMax((int*)(scal + 1), __float_as_int(red[0]));
}

__global__ __launch_bounds__(256)
void pinv_init_kernel(const float* __restrict__ a, const float* __restrict__ scal,
                      unsigned short* __restrict__ z)   // z0 = a^T / (col*row)
{
    int bh = blockIdx.x >> 8, i = blockIdx.x & 255, j = threadIdx.x;
    float inv = 1.f / (scal[0] * scal[1]);
    float v = a[(((size_t)bh * 256) + i) * 256 + j];
    z[(((size_t)bh * 256) + j) * 256 + i] = f2bf(v * inv);
}

__global__ __launch_bounds__(256)
void azw_kernel(const float* __restrict__ az, unsigned short* __restrict__ azb,
                unsigned short* __restrict__ w)   // azb = bf(az); w = 7I - az
{
    size_t idx = (size_t)blockIdx.x * 256 + threadIdx.x;
    if (idx >= (size_t)32 * 65536) return;
    int j = (int)(idx & 255), i = (int)((idx >> 8) & 255);
    float v = az[idx];
    azb[idx] = f2bf(v);
    w[idx] = f2bf(((i == j) ? 7.f : 0.f) - v);
}

__global__ __launch_bounds__(256)
void aim_kernel(const float* __restrict__ m, unsigned short* __restrict__ w, float alpha)
{
    size_t idx = (size_t)blockIdx.x * 256 + threadIdx.x;
    if (idx >= (size_t)32 * 65536) return;
    int j = (int)(idx & 255), i = (int)((idx >> 8) & 255);
    w[idx] = f2bf(((i == j) ? alpha : 0.f) - m[idx]);
}

__global__ __launch_bounds__(256)
void zscale_kernel(const float* __restrict__ zn, unsigned short* __restrict__ z)
{
    size_t idx = (size_t)blockIdx.x * 256 + threadIdx.x;
    if (idx >= (size_t)32 * 65536) return;
    z[idx] = f2bf(0.25f * zn[idx]);
}

// ---------------------------------------------------------------------------
// f32 -> bf16 conversion
// ---------------------------------------------------------------------------
__global__ __launch_bounds__(256)
void cvt_bf16_kernel(const float* __restrict__ in, unsigned short* __restrict__ out, size_t n)
{
    size_t i = (size_t)blockIdx.x * 256 + threadIdx.x;
    if (i < n) out[i] = f2bf(in[i]);
}

// ---------------------------------------------------------------------------
// Depthwise conv residual (33 taps over sequence, per head) + head-merge reshape
// ---------------------------------------------------------------------------
__global__ __launch_bounds__(256)
void conv_reshape_kernel(const float* __restrict__ outf, const float* __restrict__ vf,
                         const float* __restrict__ cw, unsigned short* __restrict__ outc)
{
    size_t idx = (size_t)blockIdx.x * 256 + threadIdx.x;   // BH*N*64
    if (idx >= (size_t)32 * 4096 * 64) return;
    int d  = (int)(idx & 63);
    int n  = (int)((idx >> 6) & 4095);
    int bh = (int)(idx >> 18);
    int h = bh & 7, b = bh >> 3;
    float acc = outf[idx];
    const float* vbase = vf + ((size_t)bh * 4096) * 64 + d;
    const float* w = cw + h * 33;
#pragma unroll
    for (int t = 0; t < 33; ++t) {
        int nn = n + t - 16;
        if (nn >= 0 && nn < 4096) acc += w[t] * vbase[(size_t)nn * 64];
    }
    outc[(((size_t)b * 4096 + n) * 512) + h * 64 + d] = f2bf(acc);
}

// ---------------------------------------------------------------------------
// y = x + attn_out + b_out
// ---------------------------------------------------------------------------
__global__ __launch_bounds__(256)
void final_add_kernel(const float* __restrict__ x, const float* __restrict__ ao,
                      const float* __restrict__ bout, float* __restrict__ y)
{
    size_t idx = (size_t)blockIdx.x * 256 + threadIdx.x;   // B*N*512
    if (idx >= (size_t)4 * 4096 * 512) return;
    y[idx] = x[idx] + ao[idx] + bout[idx & 511];
}

// ===========================================================================
extern "C" void kernel_launch(void* const* d_in, const int* in_sizes, int n_in,
                              void* d_out, int out_size, void* d_ws, size_t ws_size,
                              hipStream_t stream)
{
    (void)in_sizes; (void)n_in; (void)out_size; (void)ws_size;
    const float* x      = (const float*)d_in[0];
    const float* gamma  = (const float*)d_in[1];
    const float* beta   = (const float*)d_in[2];
    const float* w_qkv  = (const float*)d_in[3];
    const float* w_out  = (const float*)d_in[4];
    const float* b_out  = (const float*)d_in[5];
    const float* conv_w = (const float*)d_in[6];
    float* y = (float*)d_out;

    char*  ws  = (char*)d_ws;
    size_t off = 0;
    auto alloc = [&](size_t bytes) -> char* {
        char* p = ws + off;
        off += (bytes + 255) & ~(size_t)255;
        return p;
    };
    const size_t BH = 32, Nq = 4096, Dh = 64, Lm = 256, Rows = 16384;

    unsigned short* wqkv_bf = (unsigned short*)alloc(512 * 1536 * 2);
    unsigned short* wout_bf = (unsigned short*)alloc(512 * 512 * 2);
    unsigned short* xn_bf   = (unsigned short*)alloc(Rows * 512 * 2);
    float*          qkv     = (float*)alloc(Rows * 1536 * 4);
    unsigned short* qs_bf   = (unsigned short*)alloc(BH * Nq * Dh * 2);
    unsigned short* k_bf    = (unsigned short*)alloc(BH * Nq * Dh * 2);
    unsigned short* v_bf    = (unsigned short*)alloc(BH * Nq * Dh * 2);
    float*          v_f     = (float*)alloc(BH * Nq * Dh * 4);
    unsigned short* ql_bf   = (unsigned short*)alloc(BH * Lm * Dh * 2);
    unsigned short* kl_bf   = (unsigned short*)alloc(BH * Lm * Dh * 2);
    float*          Sbig    = (float*)alloc(BH * Nq * Lm * 4);   // 128 MB, reused 3x
    unsigned short* attn1_bf= (unsigned short*)alloc(BH * Nq * Lm * 2);
    unsigned short* attn3_bf= (unsigned short*)alloc(BH * Lm * Nq * 2);
    float*          S2      = (float*)alloc(BH * Lm * Lm * 4);
    unsigned short* a2_bf   = (unsigned short*)alloc(BH * Lm * Lm * 2);
    float*          a2_f    = (float*)alloc(BH * Lm * Lm * 4);
    float*          scal    = (float*)alloc(256);
    unsigned short* z_bf    = (unsigned short*)alloc(BH * Lm * Lm * 2);
    float*          az_f    = (float*)alloc(BH * Lm * Lm * 4);
    unsigned short* az_bf   = (unsigned short*)alloc(BH * Lm * Lm * 2);
    unsigned short* w_bf    = (unsigned short*)alloc(BH * Lm * Lm * 2);
    float*          m_f     = (float*)alloc(BH * Lm * Lm * 4);
    float*          zn_f    = (float*)alloc(BH * Lm * Lm * 4);
    float*          t1_f    = (float*)alloc(BH * Lm * Dh * 4);
    unsigned short* t1_bf   = (unsigned short*)alloc(BH * Lm * Dh * 2);
    unsigned short* t2_bf   = (unsigned short*)alloc(BH * Nq * Lm * 2);
    float*          out_f   = (float*)alloc(BH * Nq * Dh * 4);
    unsigned short* outc_bf = (unsigned short*)alloc(Rows * 512 * 2);
    float*          ao_f    = (float*)alloc(Rows * 512 * 4);

    // 1) weights -> bf16, layernorm
    cvt_bf16_kernel<<<(512 * 1536 + 255) / 256, 256, 0, stream>>>(w_qkv, wqkv_bf, 512 * 1536);
    cvt_bf16_kernel<<<(512 * 512 + 255) / 256, 256, 0, stream>>>(w_out, wout_bf, 512 * 512);
    layernorm_kernel<<<Rows, 256, 0, stream>>>(x, gamma, beta, xn_bf);

    // 2) qkv = xn @ w_qkv  [16384,512]x[512,1536]
    gemm_bf16_kernel<<<dim3(12, 128, 1), 256, 0, stream>>>(
        xn_bf, 0, 512, wqkv_bf, 0, 1536, 0, qkv, 0, 1536, 16384, 1536, 512);

    // 3) split heads + landmarks
    reorg_kernel<<<32768, 256, 0, stream>>>(qkv, qs_bf, k_bf, v_bf, v_f);
    landmarks_kernel<<<dim3(256, 32, 1), 64, 0, stream>>>(qkv, ql_bf, kl_bf);

    // 4) attn1 = softmax(q @ kl^T)   [4096,256] per batch
    gemm_bf16_kernel<<<dim3(2, 32, 32), 256, 0, stream>>>(
        qs_bf, Nq * Dh, 64, kl_bf, Lm * Dh, 64, 1, Sbig, Nq * Lm, 256, 4096, 256, 64);
    softmax_kernel<<<BH * Nq, 256, 0, stream>>>(Sbig, attn1_bf, nullptr, 256);

    // 5) attn2 = softmax(ql @ kl^T)  [256,256]
    gemm_bf16_kernel<<<dim3(2, 2, 32), 256, 0, stream>>>(
        ql_bf, Lm * Dh, 64, kl_bf, Lm * Dh, 64, 1, S2, Lm * Lm, 256, 256, 256, 64);
    softmax_kernel<<<BH * Lm, 256, 0, stream>>>(S2, a2_bf, a2_f, 256);

    // 6) attn3 = softmax(ql @ k^T)   [256,4096]
    gemm_bf16_kernel<<<dim3(32, 2, 32), 256, 0, stream>>>(
        ql_bf, Lm * Dh, 64, k_bf, Nq * Dh, 64, 1, Sbig, Lm * Nq, 4096, 256, 4096, 64);
    softmax_kernel<<<BH * Lm, 256, 0, stream>>>(Sbig, attn3_bf, nullptr, 4096);

    // 7) Newton-Schulz pinv of attn2 (6 iters)
    init_scalars_kernel<<<1, 32, 0, stream>>>(scal);
    rowsum_max_kernel<<<BH * Lm, 256, 0, stream>>>(a2_f, scal);
    colsum_max_kernel<<<BH * Lm, 256, 0, stream>>>(a2_f, scal);
    pinv_init_kernel<<<BH * Lm, 256, 0, stream>>>(a2_f, scal, z_bf);
    const dim3 gp(2, 2, 32);
    const int  ge = (int)((BH * Lm * Lm + 255) / 256);
    for (int it = 0; it < 6; ++it) {
        gemm_bf16_kernel<<<gp, 256, 0, stream>>>(a2_bf, 65536, 256, z_bf, 65536, 256, 0,
                                                 az_f, 65536, 256, 256, 256, 256);
        azw_kernel<<<ge, 256, 0, stream>>>(az_f, az_bf, w_bf);                 // w1 = 7I - az
        gemm_bf16_kernel<<<gp, 256, 0, stream>>>(az_bf, 65536, 256, w_bf, 65536, 256, 0,
                                                 m_f, 65536, 256, 256, 256, 256);
        aim_kernel<<<ge, 256, 0, stream>>>(m_f, w_bf, 15.f);                   // w2 = 15I - m1
        gemm_bf16_kernel<<<gp, 256, 0, stream>>>(az_bf, 65536, 256, w_bf, 65536, 256, 0,
                                                 m_f, 65536, 256, 256, 256, 256);
        aim_kernel<<<ge, 256, 0, stream>>>(m_f, w_bf, 13.f);                   // w3 = 13I - m2
        gemm_bf16_kernel<<<gp, 256, 0, stream>>>(z_bf, 65536, 256, w_bf, 65536, 256, 0,
                                                 zn_f, 65536, 256, 256, 256, 256);
        zscale_kernel<<<ge, 256, 0, stream>>>(zn_f, z_bf);                     // z = 0.25*zn
    }

    // 8) t1 = attn3 @ v   [256,64] per batch (K=4096)
    gemm_bf16_kernel<<<dim3(1, 2, 32), 256, 0, stream>>>(
        attn3_bf, Lm * Nq, 4096, v_bf, Nq * Dh, 64, 0, t1_f, Lm * Dh, 64, 256, 64, 4096);
    cvt_bf16_kernel<<<(int)((BH * Lm * Dh + 255) / 256), 256, 0, stream>>>(t1_f, t1_bf, BH * Lm * Dh);

    // 9) t2 = attn1 @ pinv   [4096,256]  (f32 into Sbig, then bf16)
    gemm_bf16_kernel<<<dim3(2, 32, 32), 256, 0, stream>>>(
        attn1_bf, Nq * Lm, 256, z_bf, 65536, 256, 0, Sbig, Nq * Lm, 256, 4096, 256, 256);
    cvt_bf16_kernel<<<(int)((BH * Nq * Lm + 255) / 256), 256, 0, stream>>>(Sbig, t2_bf, BH * Nq * Lm);

    // 10) out = t2 @ t1   [4096,64]
    gemm_bf16_kernel<<<dim3(1, 32, 32), 256, 0, stream>>>(
        t2_bf, Nq * Lm, 256, t1_bf, Lm * Dh, 64, 0, out_f, Nq * Dh, 64, 4096, 64, 256);

    // 11) + depthwise conv residual, merge heads -> bf16 [16384,512]
    conv_reshape_kernel<<<(int)((BH * Nq * Dh + 255) / 256), 256, 0, stream>>>(
        out_f, v_f, conv_w, outc_bf);

    // 12) attn_out = outc @ w_out ; y = x + attn_out + b_out
    gemm_bf16_kernel<<<dim3(4, 128, 1), 256, 0, stream>>>(
        outc_bf, 0, 512, wout_bf, 0, 512, 0, ao_f, 0, 512, 16384, 512, 512);
    final_add_kernel<<<32768, 256, 0, stream>>>(x, ao_f, b_out, y);
}
